// MultiMaxDisplacerNet_5987184411090
// MI455X (gfx1250) — matrix operand order repaired and checked by reading, not compiled, not run on hardware
//
#include <hip/hip_runtime.h>
#include <math.h>

// ---------------------------------------------------------------------------
// MultiMaxDisplacerNet on MI455X (gfx1250), wave32 + WMMA bf16 16x16x32.
// Compute-bound on X @ X^T kNN distance GEMMs (~155 GF bf16) -> all matmuls
// routed through v_wmma_f32_16x16x32_bf16 with f32 accumulation.
// ---------------------------------------------------------------------------

#define NV   4096
#define NB   3
#define KNN  16
#define CHUNKS 4          // column-chunks per row for the kNN scan
#define LDH  512          // row stride (elements) of activation buffers

typedef __bf16 bf16;
typedef __attribute__((ext_vector_type(16))) __bf16 v16bf;
typedef __attribute__((ext_vector_type(8)))  __bf16 v8bf;
typedef __attribute__((ext_vector_type(8)))  float  v8f;

// ----------------------------- WMMA tile loads -----------------------------
// A (16x32 bf16): lane L holds row M=L%16.
//   lanes 0-15 : elems 0..7 -> K 0..7,  elems 8..15 -> K 16..23
//   lanes 16-31: elems 0..7 -> K 8..15, elems 8..15 -> K 24..31
__device__ __forceinline__ v16bf load_a_tile(const bf16* __restrict__ base, int lda,
                                             int rowbase, int kbase, int lane) {
  const bf16* p = base + (long)(rowbase + (lane & 15)) * lda + kbase;
  int hi = (lane >> 4) & 1;
  v8bf lo = *(const v8bf*)(p + hi * 8);
  v8bf hh = *(const v8bf*)(p + 16 + hi * 8);
  v16bf a;
#pragma unroll
  for (int i = 0; i < 8; ++i) { a[i] = lo[i]; a[8 + i] = hh[i]; }
  return a;
}

// B (32x16 bf16): lane L holds column N=L%16.
//   lanes 0-15 : elems 0..7 -> K 0..7,  elems 8..15 -> K 16..23
//   lanes 16-31: elems 0..7 -> K 8..15, elems 8..15 -> K 24..31
__device__ __forceinline__ v16bf load_b_tile(const bf16* __restrict__ base, int ldb,
                                             int colbase, int kbase, int lane) {
  const bf16* p = base + (long)(colbase + (lane & 15)) * ldb + kbase + ((lane >> 4) & 1) * 8;
  v8bf lo = *(const v8bf*)(p);
  v8bf hh = *(const v8bf*)(p + 16);
  v16bf b;
#pragma unroll
  for (int i = 0; i < 8; ++i) { b[i] = lo[i]; b[8 + i] = hh[i]; }
  return b;
}

// Sorted (descending) top-16 insertion; fully unrolled -> register arrays.
__device__ __forceinline__ void top16_insert(float (&best)[KNN], int (&bidx)[KNN],
                                             float d, int col) {
  bool placed = false;
#pragma unroll
  for (int s = 0; s < KNN - 1; ++s) {
    if (!placed) {
      if (best[s + 1] > d) { best[s] = best[s + 1]; bidx[s] = bidx[s + 1]; }
      else                 { best[s] = d; bidx[s] = col; placed = true; }
    }
  }
  if (!placed) { best[KNN - 1] = d; bidx[KNN - 1] = col; }
}

// ---------------------- weight convert + transpose to bf16 ------------------
// W[K][N] (f32, row-major) -> WT[N][K] (bf16)
__global__ void convT_kernel(const float* __restrict__ W, bf16* __restrict__ WT,
                             int K, int N) {
  long t = (long)blockIdx.x * blockDim.x + threadIdx.x;
  if (t >= (long)K * N) return;
  int k = (int)(t / N), n = (int)(t % N);
  WT[(long)n * K + k] = (bf16)W[t];
}

// ------------------------------ feature layer -------------------------------
// h[b][v][c] = sigmoid(x[v, s:s+d] @ Wf + bf);  write f32 cat slice + bf16 buf
__global__ void feature_kernel(const float* __restrict__ x, const float* __restrict__ Wf,
                               const float* __restrict__ bfv, int s, int d, int b,
                               float* __restrict__ cat, bf16* __restrict__ hbf) {
  int t = blockIdx.x * 256 + threadIdx.x;          // 4096*256 threads
  int v = t >> 8, c = t & 255;
  float acc = bfv[c];
  for (int dd = 0; dd < d; ++dd)
    acc = fmaf(x[v * 12 + s + dd], Wf[dd * 256 + c], acc);
  float o = 1.0f / (1.0f + __expf(-acc));
  cat[((long)b * NV + v) * 2048 + c] = o;
  hbf[((long)b * NV + v) * LDH + c] = (bf16)o;
}

// ------------------------------ row sq-norms --------------------------------
__global__ void rownorm_kernel(const float* __restrict__ cat, int catOff, int C,
                               float* __restrict__ sq) {
  int v = blockIdx.x, b = blockIdx.y, lane = threadIdx.x;   // 32 threads
  const float* p = cat + ((long)b * NV + v) * 2048 + catOff;
  float s = 0.f;
  for (int c = lane; c < C; c += 32) { float val = p[c]; s = fmaf(val, val, s); }
#pragma unroll
  for (int off = 16; off > 0; off >>= 1) s += __shfl_down(s, off, 32);
  if (lane == 0) sq[b * NV + v] = s;
}

// ------------------------------ bf16 WMMA GEMM ------------------------------
// C[M][N] (f32) = A[M][K] (bf16, lda) @ BT[N][K]^T (bf16); one wave per 32x16.
__global__ void gemm_bf16_kernel(const bf16* __restrict__ A, int lda, long Abatch,
                                 const bf16* __restrict__ BT, int ldb,
                                 float* __restrict__ C, int ldc, long Cbatch,
                                 bf16* __restrict__ Cbf, int ldcbf,
                                 const float* __restrict__ bias, int relu, int K) {
  int lane = threadIdx.x & 31;
  int rowbase = blockIdx.x * 32;
  int colbase = blockIdx.y * 16;
  int b = blockIdx.z;
  const bf16* Ab = A + (long)b * Abatch;
  v8f acc0 = {}, acc1 = {};
  for (int kb = 0; kb < K; kb += 32) {
    v16bf a0 = load_a_tile(Ab, lda, rowbase,      kb, lane);
    v16bf a1 = load_a_tile(Ab, lda, rowbase + 16, kb, lane);
    v16bf bt = load_b_tile(BT, ldb, colbase,      kb, lane);
    acc0 = __builtin_amdgcn_wmma_f32_16x16x32_bf16(false, a0, false, bt, (short)0, acc0, false, false);
    acc1 = __builtin_amdgcn_wmma_f32_16x16x32_bf16(false, a1, false, bt, (short)0, acc1, false, false);
  }
  int col  = colbase + (lane & 15);
  int roff = (lane >> 4) ? 8 : 0;
  float bv = bias ? bias[col] : 0.0f;
  float* Cb = C + (long)b * Cbatch;
#pragma unroll
  for (int r = 0; r < 8; ++r) {
    int row0 = rowbase + roff + r;
    int row1 = row0 + 16;
    float v0 = acc0[r] + bv, v1 = acc1[r] + bv;
    if (relu) { v0 = fmaxf(v0, 0.f); v1 = fmaxf(v1, 0.f); }
    Cb[(long)row0 * ldc + col] = v0;
    Cb[(long)row1 * ldc + col] = v1;
    if (Cbf) {
      Cbf[(long)row0 * ldcbf + col] = (bf16)v0;
      Cbf[(long)row1 * ldcbf + col] = (bf16)v1;
    }
  }
}

// -------------------- fused WMMA distances + per-row top-16 -----------------
// One wave per (32 rows x 1024-column chunk); D = |r|^2 + |c|^2 - 2 r.c
__global__ void knn_dist_kernel(const bf16* __restrict__ X, const float* __restrict__ sq,
                                float* __restrict__ cand_d, int* __restrict__ cand_i,
                                int C) {
  __shared__ float tile[32][17];
  int lane = threadIdx.x;
  int rowbase = blockIdx.x * 32;
  int chunk   = blockIdx.y;
  int b       = blockIdx.z;
  const bf16*  Xb  = X  + (long)b * NV * LDH;
  const float* sqb = sq + (long)b * NV;

  float best[KNN]; int bidx[KNN];
#pragma unroll
  for (int s = 0; s < KNN; ++s) { best[s] = __builtin_inff(); bidx[s] = 0; }

  int grow = rowbase + lane;
  float rowsq = sqb[grow];
  int cbegin = chunk * (NV / CHUNKS);

  for (int ct = 0; ct < NV / CHUNKS; ct += 16) {
    int cbase = cbegin + ct;
    v8f acc0 = {}, acc1 = {};
    for (int kb = 0; kb < C; kb += 32) {
      v16bf a0 = load_a_tile(Xb, LDH, rowbase,      kb, lane);
      v16bf a1 = load_a_tile(Xb, LDH, rowbase + 16, kb, lane);
      v16bf bt = load_b_tile(Xb, LDH, cbase,        kb, lane);
      acc0 = __builtin_amdgcn_wmma_f32_16x16x32_bf16(false, a0, false, bt, (short)0, acc0, false, false);
      acc1 = __builtin_amdgcn_wmma_f32_16x16x32_bf16(false, a1, false, bt, (short)0, acc1, false, false);
    }
    int roff = (lane >> 4) ? 8 : 0;
    int col  = lane & 15;
#pragma unroll
    for (int r = 0; r < 8; ++r) {
      tile[roff + r][col]      = acc0[r];
      tile[16 + roff + r][col] = acc1[r];
    }
    __syncthreads();
#pragma unroll
    for (int j = 0; j < 16; ++j) {
      int gcol = cbase + j;
      float d = fmaf(-2.0f, tile[lane][j], rowsq + sqb[gcol]);
      if (gcol != grow && d < best[0]) top16_insert(best, bidx, d, gcol);
    }
    __syncthreads();
  }
  long off = (((long)b * NV + grow) * CHUNKS + chunk) * KNN;
#pragma unroll
  for (int s = 0; s < KNN; ++s) { cand_d[off + s] = best[s]; cand_i[off + s] = bidx[s]; }
}

// ------------------------- merge chunk top-16 lists -------------------------
__global__ void knn_merge_kernel(const float* __restrict__ cand_d,
                                 const int* __restrict__ cand_i,
                                 int* __restrict__ idxout) {
  int t = blockIdx.x * blockDim.x + threadIdx.x;   // b*NV + row
  if (t >= NB * NV) return;
  float best[KNN]; int bidx[KNN];
#pragma unroll
  for (int s = 0; s < KNN; ++s) { best[s] = __builtin_inff(); bidx[s] = 0; }
  for (int ch = 0; ch < CHUNKS; ++ch) {
    long off = ((long)t * CHUNKS + ch) * KNN;
#pragma unroll
    for (int s = 0; s < KNN; ++s) {
      float d = cand_d[off + s]; int ii = cand_i[off + s];
      if (d < best[0]) top16_insert(best, bidx, d, ii);
    }
  }
#pragma unroll
  for (int s = 0; s < KNN; ++s) idxout[(long)t * KNN + s] = bidx[s];
}

// ------------------- GATv2 attention softmax + aggregation ------------------
__global__ void gat_aggregate_kernel(const float* __restrict__ hl, const float* __restrict__ hr,
                                     const int* __restrict__ idx, const float* __restrict__ att,
                                     const float* __restrict__ bg,
                                     float* __restrict__ cat, int catOff,
                                     bf16* __restrict__ hbf, int Cout) {
  int n = blockIdx.x, b = blockIdx.y;
  const float* hlb = hl + (long)b * NV * LDH;
  const float* hrb = hr + (long)b * NV * LDH;
  const int*   id  = idx + ((long)b * NV + n) * KNN;
  __shared__ int   sidx[KNN];
  __shared__ float se[KNN];
  __shared__ float sred[KNN][8];
  int t = threadIdx.x;                 // 256 threads = 8 waves
  if (t < KNN) sidx[t] = id[t];
  __syncthreads();

  float ep[KNN];
#pragma unroll
  for (int k = 0; k < KNN; ++k) ep[k] = 0.f;
  for (int c = t; c < Cout; c += 256) {
    float hrv = hrb[(long)n * LDH + c];
    float av  = att[c];
#pragma unroll
    for (int k = 0; k < KNN; ++k) {
      float v = hlb[(long)sidx[k] * LDH + c] + hrv;
      v = (v > 0.f) ? v : 0.2f * v;            // leaky relu, slope 0.2
      ep[k] = fmaf(av, v, ep[k]);
    }
  }
#pragma unroll
  for (int k = 0; k < KNN; ++k) {
    float v = ep[k];
#pragma unroll
    for (int off = 16; off > 0; off >>= 1) v += __shfl_down(v, off, 32);
    ep[k] = v;
  }
  int wave = t >> 5, lane = t & 31;
  if (lane == 0) {
#pragma unroll
    for (int k = 0; k < KNN; ++k) sred[k][wave] = ep[k];
  }
  __syncthreads();
  if (t < KNN) {
    float e = 0.f;
#pragma unroll
    for (int w = 0; w < 8; ++w) e += sred[t][w];
    se[t] = e;
  }
  __syncthreads();

  float m = -__builtin_inff();
#pragma unroll
  for (int k = 0; k < KNN; ++k) m = fmaxf(m, se[k]);
  float a[KNN], den = 0.f;
#pragma unroll
  for (int k = 0; k < KNN; ++k) { a[k] = __expf(se[k] - m); den += a[k]; }
  float inv = 1.0f / den;

  for (int c = t; c < Cout; c += 256) {
    float s = 0.f;
#pragma unroll
    for (int k = 0; k < KNN; ++k) s = fmaf(a[k], hlb[(long)sidx[k] * LDH + c], s);
    float o = fmaxf(fmaf(s, inv, bg[c]), 0.f);
    cat[((long)b * NV + n) * 2048 + catOff + c] = o;
    hbf[((long)b * NV + n) * LDH + c] = (bf16)o;
  }
}

// ----------------------------- cross-batch max ------------------------------
__global__ void poolmax_kernel(const float* __restrict__ cat, bf16* __restrict__ pooled) {
  long t = (long)blockIdx.x * blockDim.x + threadIdx.x;   // NV*2048
  if (t >= (long)NV * 2048) return;
  const long bs = (long)NV * 2048;
  float m = fmaxf(fmaxf(cat[t], cat[t + bs]), cat[t + 2 * bs]);
  pooled[t] = (bf16)m;
}

// -------------------- geodesic-damped displacement head ---------------------
__global__ void geod_kernel(const float* __restrict__ y2, const float* __restrict__ Wg,
                            const float* __restrict__ bgeo, const float* __restrict__ geod,
                            float* __restrict__ out) {
  int v = blockIdx.x * blockDim.x + threadIdx.x;
  if (v >= NV) return;
  float damp = tanhf(29.444389791664407f * geod[v]);   // atanh(0.9)/0.05
#pragma unroll
  for (int j = 0; j < 3; ++j) {
    float a = bgeo[j];
    for (int c = 0; c < 64; ++c) a = fmaf(y2[(long)v * 64 + c], Wg[c * 3 + j], a);
    out[v * 3 + j] = a * damp;
  }
}

// ---------------------------------------------------------------------------
extern "C" void kernel_launch(void* const* d_in, const int* in_sizes, int n_in,
                              void* d_out, int out_size, void* d_ws, size_t ws_size,
                              hipStream_t stream) {
  (void)in_sizes; (void)n_in; (void)out_size; (void)ws_size;

  const float* x    = (const float*)d_in[0];
  const float* Wf[3]  = { (const float*)d_in[1], (const float*)d_in[3], (const float*)d_in[5] };
  const float* bf_[3] = { (const float*)d_in[2], (const float*)d_in[4], (const float*)d_in[6] };
  const float* Wl[4], *Wr[4], *att[4], *bg[4];
  for (int l = 0; l < 4; ++l) {
    Wl[l]  = (const float*)d_in[7 + 4 * l];
    Wr[l]  = (const float*)d_in[8 + 4 * l];
    att[l] = (const float*)d_in[9 + 4 * l];
    bg[l]  = (const float*)d_in[10 + 4 * l];
  }
  const float* W1   = (const float*)d_in[23];
  const float* b1   = (const float*)d_in[24];
  const float* W2   = (const float*)d_in[25];
  const float* b2   = (const float*)d_in[26];
  const float* Wg   = (const float*)d_in[27];
  const float* bgeo = (const float*)d_in[28];
  const float* geod = (const float*)d_in[29];
  float* out = (float*)d_out;

  static const int GAT_IN[4]  = {256, 256, 512, 512};
  static const int GAT_OUT[4] = {256, 512, 512, 512};
  static const int IN_OFF[4]  = {0, 256, 512, 1024};
  static const int OUT_OFF[4] = {256, 512, 1024, 1536};
  static const int FSPLIT[3]  = {0, 3, 8};
  static const int FDIM[3]    = {3, 5, 4};

  // -------- workspace carve-out --------
  char* p = (char*)d_ws;
  auto alloc = [&](size_t bytes) { void* r = (void*)p; p += (bytes + 255) & ~(size_t)255; return r; };
  bf16*  hbf    = (bf16*) alloc((size_t)NB * NV * LDH * sizeof(bf16));
  float* hlbuf  = (float*)alloc((size_t)NB * NV * LDH * sizeof(float));
  float* hrbuf  = (float*)alloc((size_t)NB * NV * LDH * sizeof(float));
  float* cat    = (float*)alloc((size_t)NB * NV * 2048 * sizeof(float));
  float* sq     = (float*)alloc((size_t)NB * NV * sizeof(float));
  float* cand_d = (float*)alloc((size_t)NB * NV * CHUNKS * KNN * sizeof(float));
  int*   cand_i = (int*)  alloc((size_t)NB * NV * CHUNKS * KNN * sizeof(int));
  int*   idxbuf = (int*)  alloc((size_t)NB * NV * KNN * sizeof(int));
  bf16*  pooled = (bf16*) alloc((size_t)NV * 2048 * sizeof(bf16));
  float* y1     = (float*)alloc((size_t)NV * 256 * sizeof(float));
  bf16*  y1bf   = (bf16*) alloc((size_t)NV * 256 * sizeof(bf16));
  float* y2     = (float*)alloc((size_t)NV * 64 * sizeof(float));
  bf16*  WlT[4], *WrT[4];
  for (int l = 0; l < 4; ++l) {
    WlT[l] = (bf16*)alloc((size_t)GAT_IN[l] * GAT_OUT[l] * sizeof(bf16));
    WrT[l] = (bf16*)alloc((size_t)GAT_IN[l] * GAT_OUT[l] * sizeof(bf16));
  }
  bf16* W1T = (bf16*)alloc((size_t)2048 * 256 * sizeof(bf16));
  bf16* W2T = (bf16*)alloc((size_t)256 * 64 * sizeof(bf16));

  // -------- weight convert+transpose to bf16 [N][K] --------
  auto convT = [&](const float* W, bf16* WT, int K, int N) {
    long n = (long)K * N;
    convT_kernel<<<dim3((unsigned)((n + 255) / 256)), 256, 0, stream>>>(W, WT, K, N);
  };
  for (int l = 0; l < 4; ++l) {
    convT(Wl[l], WlT[l], GAT_IN[l], GAT_OUT[l]);
    convT(Wr[l], WrT[l], GAT_IN[l], GAT_OUT[l]);
  }
  convT(W1, W1T, 2048, 256);
  convT(W2, W2T, 256, 64);

  // -------- feature transform (sigmoid) --------
  for (int b = 0; b < NB; ++b)
    feature_kernel<<<dim3(NV), 256, 0, stream>>>(x, Wf[b], bf_[b], FSPLIT[b], FDIM[b], b, cat, hbf);

  // -------- 4 GATv2 blocks --------
  const long AB = (long)NV * LDH;
  for (int l = 0; l < 4; ++l) {
    int Cin = GAT_IN[l], Cout = GAT_OUT[l];
    rownorm_kernel<<<dim3(NV, NB), 32, 0, stream>>>(cat, IN_OFF[l], Cin, sq);
    gemm_bf16_kernel<<<dim3(NV / 32, Cout / 16, NB), 32, 0, stream>>>(
        hbf, LDH, AB, WlT[l], Cin, hlbuf, LDH, AB, (bf16*)nullptr, 0, nullptr, 0, Cin);
    gemm_bf16_kernel<<<dim3(NV / 32, Cout / 16, NB), 32, 0, stream>>>(
        hbf, LDH, AB, WrT[l], Cin, hrbuf, LDH, AB, (bf16*)nullptr, 0, nullptr, 0, Cin);
    knn_dist_kernel<<<dim3(NV / 32, CHUNKS, NB), 32, 0, stream>>>(hbf, sq, cand_d, cand_i, Cin);
    knn_merge_kernel<<<dim3((NB * NV + 255) / 256), 256, 0, stream>>>(cand_d, cand_i, idxbuf);
    gat_aggregate_kernel<<<dim3(NV, NB), 256, 0, stream>>>(
        hlbuf, hrbuf, idxbuf, att[l], bg[l], cat, OUT_OFF[l], hbf, Cout);
  }

  // -------- max-pool over batches, then MLP head --------
  poolmax_kernel<<<dim3((unsigned)(((long)NV * 2048 + 255) / 256)), 256, 0, stream>>>(cat, pooled);
  gemm_bf16_kernel<<<dim3(NV / 32, 256 / 16, 1), 32, 0, stream>>>(
      pooled, 2048, 0, W1T, 2048, y1, 256, 0, y1bf, 256, b1, 1, 2048);
  gemm_bf16_kernel<<<dim3(NV / 32, 64 / 16, 1), 32, 0, stream>>>(
      y1bf, 256, 0, W2T, 256, y2, 64, 0, (bf16*)nullptr, 0, b2, 1, 256);
  geod_kernel<<<dim3(NV / 256), 256, 0, stream>>>(y2, Wg, bgeo, geod, out);
}
